// SourceFilterFormantSynthesisGenerator_16200616640875
// MI455X (gfx1250) — compile-verified, matrix operand order checked by reading, not run on hardware
//
#include <hip/hip_runtime.h>
#include <hip/hip_bf16.h>

typedef __attribute__((ext_vector_type(16))) _Float16 v16h;
typedef __attribute__((ext_vector_type(8)))  _Float16 v8h;
typedef __attribute__((ext_vector_type(4)))  _Float16 v4h;
typedef __attribute__((ext_vector_type(8)))  float    v8f;

// ---------------------------------------------------------------------------
// Problem constants
// ---------------------------------------------------------------------------
#define NB      16
#define NFEAT   16
#define NFRAMES 128
#define ORDER   30
#define NMELS   80
#define NOUT    111      // ORDER + 1 + NMELS
#define RES     128
#define SKIPC   128
#define NLAY    8
#define HOP     256
#define NFFT    512
#define NBINS   257
#define TSAMP   32768    // 128 * 8*8*2*2

// d_out layout (floats): y | H | mel
#define OUT_Y   0
#define OUT_H   (NB * TSAMP)                       // 524288
#define OUT_MEL (OUT_H + NB * NBINS * NFRAMES)     // 1050624

// workspace layout (floats)
#define WS_H       ((size_t)0)                     // 16*128*128
#define WS_Z       (WS_H + 262144)                 // 16*256*128
#define WS_G       (WS_Z + 524288)
#define WS_SKIP    (WS_G + 262144)
#define WS_XOUT    (WS_SKIP + 262144)              // 16*111*128 (padded)
#define WS_GANIN   (WS_XOUT + 229376)
#define WS_ALLPOLE (WS_GANIN + 229376)             // 16*31*128 (padded)
#define WS_GAIN    (WS_ALLPOLE + 65536)            // 16*128
#define WS_B0      (WS_GAIN + 2048)                // 8388608
#define WS_B1      (WS_B0 + 8388608)               // 8388608 (also excitation)
#define WS_B2      (WS_B1 + 8388608)               // 8388608

// ---------------------------------------------------------------------------
// WMMA fragment K-pattern (ISA 7.12.2, 16-bit A 16x32): fragment element i:
// i<8 -> K = i + 8*half ; i>=8 -> K = (i-8) + 16 + 8*half  (two 8-runs)
// ---------------------------------------------------------------------------
__device__ __forceinline__ int kpat(int v, int e, int half) {
  return 2 * (v & 3) + ((v >> 2) << 4) + (half << 3) + e;
}

template<int ACT>
__device__ __forceinline__ float act_apply(float x) {
  if (ACT == 1) return x > 0.f ? x : 0.1f * x;   // leaky relu 0.1
  if (ACT == 2) return x > 0.f ? x : 0.f;        // relu
  return x;
}

// Load A fragment (f32 weights -> f16) : two contiguous runs of 8 floats.
__device__ __forceinline__ v16h load_a_frag(const float* __restrict__ Wrow,
                                            int kk0, int KK, int half,
                                            bool rowvalid, bool vec_ok) {
  v16h a;
  if (rowvalid && vec_ok && (kk0 + 32 <= KK)) {
    const float4* wp = (const float4*)(Wrow + kk0);
    float4 w0 = wp[2 * half];
    float4 w1 = wp[2 * half + 1];
    float4 w2 = wp[4 + 2 * half];
    float4 w3 = wp[5 + 2 * half];
    a[0] = (_Float16)w0.x;  a[1] = (_Float16)w0.y;  a[2] = (_Float16)w0.z;  a[3] = (_Float16)w0.w;
    a[4] = (_Float16)w1.x;  a[5] = (_Float16)w1.y;  a[6] = (_Float16)w1.z;  a[7] = (_Float16)w1.w;
    a[8] = (_Float16)w2.x;  a[9] = (_Float16)w2.y;  a[10] = (_Float16)w2.z; a[11] = (_Float16)w2.w;
    a[12] = (_Float16)w3.x; a[13] = (_Float16)w3.y; a[14] = (_Float16)w3.z; a[15] = (_Float16)w3.w;
  } else {
#pragma unroll
    for (int v = 0; v < 8; ++v)
#pragma unroll
      for (int e = 0; e < 2; ++e) {
        int kk = kk0 + kpat(v, e, half);
        a[2 * v + e] = (rowvalid && kk < KK) ? (_Float16)Wrow[kk] : (_Float16)0.f;
      }
  }
  return a;
}

// Load B fragment from column-major LDS tile (col = 32 contiguous halves).
__device__ __forceinline__ v16h load_b_frag(const _Float16* __restrict__ colp, int half) {
  union { v16h v; v8h h[2]; } u;
  u.h[0] = *(const v8h*)(colp + 8 * half);        // ds_load_b128
  u.h[1] = *(const v8h*)(colp + 16 + 8 * half);   // ds_load_b128
  return u.v;
}

// ---------------------------------------------------------------------------
// Dilated conv1d as WMMA GEMM. Block = (32,4): 4 waves = 4 M-subtiles sharing
// one N-tile; cooperative 32x16 B tile in LDS (staged 4 rows/thread, b64
// packed stores). grid = (ceil(Cout/64), T/16, B)
// ---------------------------------------------------------------------------
template<int KT, int ACT>
__global__ void conv1d_wmma(const float* __restrict__ X, const float* __restrict__ W,
                            const float* __restrict__ Bias, float* __restrict__ Y,
                            int Cin, int Cout, int T, int dil,
                            int relu_out, int accumulate) {
  const int lane = threadIdx.x;
  const int wid  = threadIdx.y;
  const int tid  = wid * 32 + lane;
  const int m0   = (blockIdx.x * 4 + wid) * 16;
  const int n0   = blockIdx.y * 16;
  const int b    = blockIdx.z;

  const int pad = dil * (KT - 1) / 2;
  const int KK  = Cin * KT;
  const bool vec_ok = ((KK & 3) == 0);
  const float* __restrict__ Xb = X + b * Cin * T;     // 32-bit inner indexing

  __shared__ __align__(16) _Float16 Bs[16][32];       // [col][kk]

  // staging: thread -> column j, 4 consecutive kk rows
  const int j  = tid & 15;
  const int kb = (tid >> 4) * 4;
  const int t  = n0 + j;
  const bool interior = (n0 - pad >= 0) && (n0 + 15 + (KT - 1) * dil - pad < T);

  const int row  = lane & 15;
  const int half = lane >> 4;
  const int co_a = m0 + row;
  const bool rowvalid = (co_a < Cout);
  const float* __restrict__ Wrow = W + (size_t)co_a * KK;
  const _Float16* colp = &Bs[row][0];

  v8f creg = {};
  for (int kk0 = 0; kk0 < KK; kk0 += 32) {
    union { v4h v; _Float16 s[4]; } pack;
    if (interior && (kk0 + 32 <= KK)) {
      // fast path: no per-lane guards, no exec churn
#pragma unroll
      for (int q = 0; q < 4; ++q) {
        int kk = kk0 + kb + q;
        int ci = kk / KT, tap = kk - ci * KT;
        pack.s[q] = (_Float16)act_apply<ACT>(Xb[ci * T + t + tap * dil - pad]);
      }
    } else {
#pragma unroll
      for (int q = 0; q < 4; ++q) {
        int kk = kk0 + kb + q;
        float val = 0.f;
        if (kk < KK) {
          int ci = kk / KT, tap = kk - ci * KT;
          int ti = t + tap * dil - pad;
          if (ti >= 0 && ti < T) val = act_apply<ACT>(Xb[ci * T + ti]);
        }
        pack.s[q] = (_Float16)val;
      }
    }
    *(v4h*)(&Bs[j][kb]) = pack.v;                     // ds_store_b64
    __syncthreads();

    if (rowvalid && kk0 + 32 < KK) __builtin_prefetch(Wrow + kk0 + 32, 0, 1);
    v16h afr = load_a_frag(Wrow, kk0, KK, half, rowvalid, vec_ok);
    v16h bfr = load_b_frag(colp, half);
    creg = __builtin_amdgcn_wmma_f32_16x16x32_f16(false, afr, false, bfr,
                                                  (short)0, creg, false, false);
    __syncthreads();
  }

  // store D: lane<16 -> M=r, N=lane ; lane>=16 -> M=r+8, N=lane-16
  const int n = n0 + (lane & 15);
  const int mbase = m0 + 8 * (lane >> 4);
#pragma unroll
  for (int r = 0; r < 8; ++r) {
    int co = mbase + r;
    if (co < Cout) {
      float val = creg[r] + Bias[co];
      if (relu_out) val = fmaxf(val, 0.f);
      int idx = (b * Cout + co) * T + n;
      if (accumulate) Y[idx] += val; else Y[idx] = val;
    }
  }
}

// ---------------------------------------------------------------------------
// Transposed conv1d (upsampling), one residue class per grid.z slice so the
// tap set is column-invariant; KK = 2*CIN is always a multiple of 32.
// grid = (ceil(Cout/64), Tin/16, B*STRIDE) ; block = (32,4)
// ---------------------------------------------------------------------------
template<int CIN, int STRIDE, int KT>
__global__ void convT1d_wmma(const float* __restrict__ X, const float* __restrict__ W,
                             const float* __restrict__ Bias, float* __restrict__ Y,
                             int Cout, int Tin) {
  const int lane = threadIdx.x;
  const int wid  = threadIdx.y;
  const int tid  = wid * 32 + lane;
  const int b    = blockIdx.z / STRIDE;
  const int res  = blockIdx.z % STRIDE;
  const int m0   = (blockIdx.x * 4 + wid) * 16;
  const int q0   = blockIdx.y * 16;               // output t = (q0+j)*STRIDE+res

  const int p    = KT - 1 - (KT - STRIDE) / 2;
  const int kr   = ((p - res) % STRIDE + STRIDE) % STRIDE;
  const int ntap = (KT - kr + STRIDE - 1) / STRIDE;    // == 2 for all stages
  const int base = (res - p + kr) / STRIDE;            // exact, may be negative
  const int KK   = CIN * ntap;
  const float* __restrict__ Xb = X + b * CIN * Tin;

  __shared__ __align__(16) _Float16 Bs[16][32];

  const int j  = tid & 15;
  const int kb = (tid >> 4) * 4;
  const bool interior = (q0 + base >= 0) && (q0 + 15 + (ntap - 1) + base < Tin);

  const int row  = lane & 15;
  const int half = lane >> 4;
  const int co_a = m0 + row;
  const bool rowvalid = (co_a < Cout);
  const _Float16* colp = &Bs[row][0];

  v8f creg = {};
  for (int kk0 = 0; kk0 < KK; kk0 += 32) {
    union { v4h v; _Float16 s[4]; } pack;
    if (interior) {
#pragma unroll
      for (int q = 0; q < 4; ++q) {
        int kk = kk0 + kb + q;
        int tap = kk / CIN, ci = kk & (CIN - 1);
        float x = Xb[ci * Tin + q0 + j + tap + base];
        pack.s[q] = (_Float16)(x > 0.f ? x : 0.1f * x);
      }
    } else {
#pragma unroll
      for (int q = 0; q < 4; ++q) {
        int kk = kk0 + kb + q;
        int tap = kk / CIN, ci = kk & (CIN - 1);
        int i = q0 + j + tap + base;
        float val = 0.f;
        if (i >= 0 && i < Tin) {
          float x = Xb[ci * Tin + i];
          val = x > 0.f ? x : 0.1f * x;
        }
        pack.s[q] = (_Float16)val;
      }
    }
    *(v4h*)(&Bs[j][kb]) = pack.v;
    __syncthreads();

    v16h afr;                                         // scalar gathers (strided)
#pragma unroll
    for (int v = 0; v < 8; ++v)
#pragma unroll
      for (int e = 0; e < 2; ++e) {
        int kk = kk0 + kpat(v, e, half);              // KK % 32 == 0: no tail
        int tap = kk / CIN, ci = kk & (CIN - 1);
        int k = kr + tap * STRIDE;                    // flipped kernel index
        float wv = rowvalid ? W[((size_t)co_a * CIN + ci) * KT + (KT - 1 - k)] : 0.f;
        afr[2 * v + e] = (_Float16)wv;
      }
    v16h bfr = load_b_frag(colp, half);
    creg = __builtin_amdgcn_wmma_f32_16x16x32_f16(false, afr, false, bfr,
                                                  (short)0, creg, false, false);
    __syncthreads();
  }

  const int col = lane & 15;
  const int mbase = m0 + 8 * (lane >> 4);
  const int Tout = Tin * STRIDE;
#pragma unroll
  for (int r = 0; r < 8; ++r) {
    int co = mbase + r;
    int t  = (q0 + col) * STRIDE + res;
    if (co < Cout)
      Y[(b * Cout + co) * Tout + t] = creg[r] + Bias[co];
  }
}

// ---------------------------------------------------------------------------
// small elementwise kernels
// ---------------------------------------------------------------------------
__global__ void fill_zero(float* __restrict__ p, int n) {
  int i = blockIdx.x * blockDim.x + threadIdx.x;
  if (i < n) p[i] = 0.f;
}

__global__ void gate_kernel(const float* __restrict__ Z, float* __restrict__ G) {
  int i = blockIdx.x * blockDim.x + threadIdx.x;        // NB*RES*NFRAMES
  if (i >= NB * RES * NFRAMES) return;
  int b = i / (RES * NFRAMES);
  int rem = i - b * RES * NFRAMES;
  int c = rem / NFRAMES, t = rem % NFRAMES;
  float zt = Z[(b * 2 * RES + c) * NFRAMES + t];
  float zs = Z[(b * 2 * RES + RES + c) * NFRAMES + t];
  G[i] = tanhf(zt) * (1.f / (1.f + expf(-zs)));
}

__global__ void mel_copy(const float* __restrict__ Xo, float* __restrict__ gan_in,
                         float* __restrict__ mel_out) {
  int i = blockIdx.x * blockDim.x + threadIdx.x;        // NB*NMELS*NFRAMES
  if (i >= NB * NMELS * NFRAMES) return;
  int b = i / (NMELS * NFRAMES);
  int rem = i - b * NMELS * NFRAMES;
  int c = rem / NFRAMES, t = rem % NFRAMES;
  float v = Xo[(b * NOUT + ORDER + 1 + c) * NFRAMES + t];
  gan_in[(b * NOUT + c) * NFRAMES + t] = v;             // channels 0..79
  mel_out[i] = v;
}

// per (b,t): rc = tanh(0.5*lars), gain = exp(xg+1e-5), forward Levinson
__global__ void levinson_kernel(const float* __restrict__ Xo, float* __restrict__ gan_in,
                                float* __restrict__ allpole, float* __restrict__ gain) {
  int i = blockIdx.x * blockDim.x + threadIdx.x;        // NB*NFRAMES
  if (i >= NB * NFRAMES) return;
  int b = i / NFRAMES, t = i % NFRAMES;
  const float* xb = Xo + b * NOUT * NFRAMES;

  float rc[ORDER], a[ORDER + 1], na[ORDER + 1];
#pragma unroll 1
  for (int m = 0; m < ORDER; ++m) {
    float l = xb[m * NFRAMES + t];
    rc[m] = tanhf(0.5f * l);
    gan_in[(b * NOUT + NMELS + m) * NFRAMES + t] = rc[m];
  }
  float xg = xb[ORDER * NFRAMES + t];
  gan_in[(b * NOUT + NMELS + ORDER) * NFRAMES + t] = xg;
  float g = expf(xg + 1e-5f);
  gain[i] = g;

  a[0] = 1.f;
  for (int jq = 1; jq <= ORDER; ++jq) a[jq] = 0.f;
#pragma unroll 1
  for (int m = 0; m < ORDER; ++m) {
    float k = rc[m];
    for (int jq = 0; jq <= m + 1; ++jq) na[jq] = a[jq] + k * a[m + 1 - jq];
    for (int jq = 0; jq <= m + 1; ++jq) a[jq] = na[jq];
  }
  for (int n = 0; n <= ORDER; ++n)
    allpole[(b * (ORDER + 1) + n) * NFRAMES + t] = a[n];
}

// H[b,f,t] = gain / (|DFT_512(allpole)[f]| + 1e-6)
__global__ void dft_h_kernel(const float* __restrict__ allpole, const float* __restrict__ gain,
                             float* __restrict__ H) {
  int i = blockIdx.x * blockDim.x + threadIdx.x;        // NB*NBINS*NFRAMES
  if (i >= NB * NBINS * NFRAMES) return;
  int b = i / (NBINS * NFRAMES);
  int rem = i - b * NBINS * NFRAMES;
  int f = rem / NFRAMES, t = rem % NFRAMES;
  float w0 = -2.f * 3.14159265358979323846f * (float)f / (float)NFFT;
  float wr = 0.f, wi = 0.f;
#pragma unroll 1
  for (int n = 0; n <= ORDER; ++n) {
    float an = allpole[(b * (ORDER + 1) + n) * NFRAMES + t];
    float s, c;
    __sincosf(w0 * (float)n, &s, &c);
    wr += an * c; wi += an * s;
  }
  float A = sqrtf(wr * wr + wi * wi);
  H[i] = gain[b * NFRAMES + t] / (A + 1e-6f);
}

// Serial order-30 IIR: one wave32 per batch element, history in lanes,
// dot product by shfl_xor reduction, history shift by shfl_up.
__global__ void synth_kernel(const float* __restrict__ exc, const float* __restrict__ allpole,
                             const float* __restrict__ gain, float* __restrict__ y) {
  const int b = blockIdx.x;
  const int lane = threadIdx.x;
  float c = 0.f, a_l = 0.f, g = 0.f;
#pragma unroll 1
  for (int t = 0; t < TSAMP; ++t) {
    if ((t & (HOP - 1)) == 0) {
      int frame = t >> 8;
      a_l = (lane < ORDER) ? allpole[(b * (ORDER + 1) + 1 + lane) * NFRAMES + frame] : 0.f;
      g = gain[b * NFRAMES + frame];
    }
    float e = tanhf(exc[b * TSAMP + t]) * g;
    float s = a_l * c;
#pragma unroll
    for (int off = 16; off > 0; off >>= 1) s += __shfl_xor(s, off, 32);
    float yn = e - s;
    float cprev = __shfl_up(c, 1, 32);
    c = (lane == 0) ? yn : cprev;
    if (lane == 0) y[b * TSAMP + t] = yn;
  }
}

// ---------------------------------------------------------------------------
// host side
// ---------------------------------------------------------------------------
static inline void launch_conv(hipStream_t s, const float* X, const float* W, const float* Bi,
                               float* Y, int Cin, int Cout, int T, int Kt, int dil,
                               int act, int relu_out, int acc) {
  dim3 grid((Cout + 63) / 64, (T + 15) / 16, NB);
  dim3 block(32, 4, 1);
#define LC(kt, a) conv1d_wmma<kt, a><<<grid, block, 0, s>>>(X, W, Bi, Y, Cin, Cout, T, dil, relu_out, acc)
  if      (Kt == 1 && act == 0) LC(1, 0);
  else if (Kt == 1 && act == 2) LC(1, 2);
  else if (Kt == 3 && act == 0) LC(3, 0);
  else if (Kt == 3 && act == 1) LC(3, 1);
  else if (Kt == 7 && act == 0) LC(7, 0);
  else if (Kt == 7 && act == 1) LC(7, 1);
#undef LC
}

static inline void launch_convT(hipStream_t s, const float* X, const float* W, const float* Bi,
                                float* Y, int Cin, int Cout, int Tin, int stride, int Kt) {
  dim3 grid((Cout + 63) / 64, (Tin + 15) / 16, NB * stride);
  dim3 block(32, 4, 1);
#define LT(cin, st, kt) convT1d_wmma<cin, st, kt><<<grid, block, 0, s>>>(X, W, Bi, Y, Cout, Tin)
  if      (Cin == 256 && stride == 8 && Kt == 16) LT(256, 8, 16);
  else if (Cin == 128 && stride == 8 && Kt == 16) LT(128, 8, 16);
  else if (Cin == 64  && stride == 2 && Kt == 4)  LT(64, 2, 4);
  else if (Cin == 32  && stride == 2 && Kt == 4)  LT(32, 2, 4);
#undef LT
}

extern "C" void kernel_launch(void* const* d_in, const int* in_sizes, int n_in,
                              void* d_out, int out_size, void* d_ws, size_t ws_size,
                              hipStream_t stream) {
  (void)in_sizes; (void)n_in; (void)out_size; (void)ws_size;
  const float* feat   = (const float*)d_in[0];
  const float* in_w   = (const float*)d_in[1];
  const float* in_b   = (const float*)d_in[2];
  const float* dil_w  = (const float*)d_in[3];
  const float* dil_b  = (const float*)d_in[4];
  const float* res_w  = (const float*)d_in[5];
  const float* res_b  = (const float*)d_in[6];
  const float* skip_w = (const float*)d_in[7];
  const float* skip_b = (const float*)d_in[8];
  const float* out1_w = (const float*)d_in[9];
  const float* out1_b = (const float*)d_in[10];
  const float* out2_w = (const float*)d_in[11];
  const float* out2_b = (const float*)d_in[12];
  const float* pre_w  = (const float*)d_in[13];
  const float* pre_b  = (const float*)d_in[14];
  const float* post_w = (const float*)d_in[15];
  const float* post_b = (const float*)d_in[16];
  const float* up_w[4]  = {(const float*)d_in[17], (const float*)d_in[21], (const float*)d_in[25], (const float*)d_in[29]};
  const float* up_b[4]  = {(const float*)d_in[18], (const float*)d_in[22], (const float*)d_in[26], (const float*)d_in[30]};
  const float* rb_w[4]  = {(const float*)d_in[19], (const float*)d_in[23], (const float*)d_in[27], (const float*)d_in[31]};
  const float* rb_b[4]  = {(const float*)d_in[20], (const float*)d_in[24], (const float*)d_in[28], (const float*)d_in[32]};

  float* ws = (float*)d_ws;
  float* out = (float*)d_out;

  float* h       = ws + WS_H;
  float* z       = ws + WS_Z;
  float* gbuf    = ws + WS_G;
  float* skip    = ws + WS_SKIP;
  float* xout    = ws + WS_XOUT;
  float* ganin   = ws + WS_GANIN;
  float* allpole = ws + WS_ALLPOLE;
  float* gain    = ws + WS_GAIN;
  float* b0      = ws + WS_B0;
  float* b1      = ws + WS_B1;
  float* b2      = ws + WS_B2;
  float* exc     = b1;                              // b1 is dead by post-conv time

  // ---------------- WaveNet feature model ----------------
  fill_zero<<<(NB * SKIPC * NFRAMES + 255) / 256, 256, 0, stream>>>(skip, NB * SKIPC * NFRAMES);
  launch_conv(stream, feat, in_w, in_b, h, NFEAT, RES, NFRAMES, 1, 1, 0, 0, 0);
  static const int DILS[NLAY] = {1, 2, 4, 8, 16, 32, 64, 128};
  for (int i = 0; i < NLAY; ++i) {
    launch_conv(stream, h, dil_w + (size_t)i * 2 * RES * RES * 3, dil_b + i * 2 * RES,
                z, RES, 2 * RES, NFRAMES, 3, DILS[i], 0, 0, 0);
    gate_kernel<<<(NB * RES * NFRAMES + 255) / 256, 256, 0, stream>>>(z, gbuf);
    launch_conv(stream, gbuf, skip_w + (size_t)i * SKIPC * RES, skip_b + i * SKIPC,
                skip, RES, SKIPC, NFRAMES, 1, 1, 0, 0, 1);
    launch_conv(stream, gbuf, res_w + (size_t)i * RES * RES, res_b + i * RES,
                h, RES, RES, NFRAMES, 1, 1, 0, 0, 1);
  }
  // o = relu(skip); o1 = relu(conv(o)); xout = conv(o1)
  launch_conv(stream, skip, out1_w, out1_b, z, SKIPC, SKIPC, NFRAMES, 1, 1, /*relu in*/2, /*relu out*/1, 0);
  launch_conv(stream, z, out2_w, out2_b, xout, SKIPC, NOUT, NFRAMES, 1, 1, 0, 0, 0);

  // ---------------- LPC head: Levinson, DFT magnitude, H -------------
  levinson_kernel<<<(NB * NFRAMES + 255) / 256, 256, 0, stream>>>(xout, ganin, allpole, gain);
  mel_copy<<<(NB * NMELS * NFRAMES + 255) / 256, 256, 0, stream>>>(xout, ganin, out + OUT_MEL);
  dft_h_kernel<<<(NB * NBINS * NFRAMES + 255) / 256, 256, 0, stream>>>(allpole, gain, out + OUT_H);

  // ---------------- HiFi generator ----------------
  launch_conv(stream, ganin, pre_w, pre_b, b0, NOUT, 256, NFRAMES, 7, 1, 0, 0, 0);

  static const int CH[5]  = {256, 128, 64, 32, 16};
  static const int UPR[4] = {8, 8, 2, 2};
  static const int UPK[4] = {16, 16, 4, 4};
  static const int RBD[3] = {1, 3, 5};
  float* cur = b0;
  float* nxt = b1;
  int T = NFRAMES;
  for (int s = 0; s < 4; ++s) {
    launch_convT(stream, cur, up_w[s], up_b[s], nxt, CH[s], CH[s + 1], T, UPR[s], UPK[s]);
    T *= UPR[s];
    int C = CH[s + 1];
    for (int j = 0; j < 3; ++j) {
      launch_conv(stream, nxt, rb_w[s] + (size_t)(2 * j) * C * C * 3, rb_b[s] + (2 * j) * C,
                  b2, C, C, T, 3, RBD[j], /*lrelu in*/1, 0, 0);
      launch_conv(stream, b2, rb_w[s] + (size_t)(2 * j + 1) * C * C * 3, rb_b[s] + (2 * j + 1) * C,
                  nxt, C, C, T, 3, 1, /*lrelu in*/1, 0, /*acc*/1);
    }
    float* tmp = cur; cur = nxt; nxt = tmp;
  }
  // cur now holds stage-3 output [16, 16, 32768]
  launch_conv(stream, cur, post_w, post_b, exc, CH[4], 1, TSAMP, 7, 1, /*lrelu in*/1, 0, 0);

  // ---------------- synthesis filter (tanh folded in) ----------------
  synth_kernel<<<NB, 32, 0, stream>>>(exc, allpole, gain, out + OUT_Y);
}
